// MASSGate_41738492183161
// MI455X (gfx1250) — matrix operand order repaired and checked by reading, not compiled, not run on hardware
//
#include <hip/hip_runtime.h>
#include <hip/hip_bf16.h>
#include <stdint.h>

// ---------------------------------------------------------------------------
// MASS gate: scores = softmax(mask(x @ W^T)) + 1e-14 ; adaptive top-k vs 0.5
// GEMM via bf16 WMMA, fp32 hi/lo split (3 WMMAs per tile -> ~fp32 accuracy).
// v2: KC=64 + double-buffered LDS staging (1 barrier per chunk, 64 total),
//     hardware bf16 converts via __builtin_convertvector.
// ---------------------------------------------------------------------------

typedef __attribute__((ext_vector_type(16))) __bf16 v16bf;
typedef __attribute__((ext_vector_type(4)))  __bf16 bf4;
typedef __attribute__((ext_vector_type(8)))  float  v8f;
typedef __attribute__((ext_vector_type(4)))  float  fvec4;

#define NUM_TOKENS 16384
#define MODEL_DIM  4096
#define MAX_E      128
#define TOK_PER_WG 64
#define KC         64
#define NKC        (MODEL_DIM / KC)   // 64 k-chunks
#define THRESH     0.5f

union Frag {
    v16bf v;
    uint4 q[2];
};

__global__ __launch_bounds__(256)
void mass_gate_kernel(const float* __restrict__ x,
                      const float* __restrict__ wg,
                      const float* __restrict__ emask,
                      float* __restrict__ out) {
    // double-buffered bf16 hi/lo staging tiles
    __shared__ __bf16 sAhi[2][TOK_PER_WG * KC];   // 2 x 8 KB
    __shared__ __bf16 sAlo[2][TOK_PER_WG * KC];   // 2 x 8 KB
    __shared__ __bf16 sBhi[2][MAX_E * KC];        // 2 x 16 KB
    __shared__ __bf16 sBlo[2][MAX_E * KC];        // 2 x 16 KB
    __shared__ float  sScore[TOK_PER_WG * MAX_E]; // 32 KB
    __shared__ float  sMask[MAX_E];

    const int tid     = threadIdx.x;
    const int lane    = tid & 31;
    const int wave    = tid >> 5;
    const int tokBase = blockIdx.x * TOK_PER_WG;

    if (tid < MAX_E) sMask[tid] = emask[tid];

    // wave tiling: 4 M-blocks (16 tokens) x 2 N-halves (4 expert tiles each)
    const int mB = wave & 3;   // token block index
    const int nH = wave >> 2;  // expert half (0 -> experts 0..63, 1 -> 64..127)

    v8f acc[4];
#pragma unroll
    for (int t = 0; t < 4; ++t)
        acc[t] = v8f{0.f, 0.f, 0.f, 0.f, 0.f, 0.f, 0.f, 0.f};

    fvec4 ra[4];   // A prefetch: 64 rows x 64 cols = 1024 f4 / 256 thr
    fvec4 rb[8];   // B prefetch: 128 rows x 64 cols = 2048 f4 / 256 thr

    auto loadChunk = [&](int kc) {
        const int kOff = kc * KC;
#pragma unroll
        for (int j = 0; j < 4; ++j) {
            int i = tid + 256 * j, row = i >> 4, c4 = i & 15;
            ra[j] = *(const fvec4*)(x + (size_t)(tokBase + row) * MODEL_DIM + kOff + c4 * 4);
        }
#pragma unroll
        for (int j = 0; j < 8; ++j) {
            int i = tid + 256 * j, row = i >> 4, c4 = i & 15;
            rb[j] = *(const fvec4*)(wg + (size_t)row * MODEL_DIM + kOff + c4 * 4);
        }
    };

    auto cvtStore = [&](int buf) {
#pragma unroll
        for (int j = 0; j < 4; ++j) {
            int i = tid + 256 * j, row = i >> 4, c4 = i & 15;
            bf4   h  = __builtin_convertvector(ra[j], bf4);      // v_cvt_pk_bf16_f32
            fvec4 hf = __builtin_convertvector(h, fvec4);
            bf4   l  = __builtin_convertvector(ra[j] - hf, bf4);
            *(bf4*)&sAhi[buf][row * KC + c4 * 4] = h;
            *(bf4*)&sAlo[buf][row * KC + c4 * 4] = l;
        }
#pragma unroll
        for (int j = 0; j < 8; ++j) {
            int i = tid + 256 * j, row = i >> 4, c4 = i & 15;
            bf4   h  = __builtin_convertvector(rb[j], bf4);
            fvec4 hf = __builtin_convertvector(h, fvec4);
            bf4   l  = __builtin_convertvector(rb[j] - hf, bf4);
            *(bf4*)&sBhi[buf][row * KC + c4 * 4] = h;
            *(bf4*)&sBlo[buf][row * KC + c4 * 4] = l;
        }
    };

    // prologue: fill buffer 0
    loadChunk(0);
    cvtStore(0);
    __syncthreads();

    for (int kc = 0; kc < NKC; ++kc) {
        const int buf = kc & 1;

        if (kc + 1 < NKC) loadChunk(kc + 1);   // global reads overlap WMMA below

#pragma unroll
        for (int kk = 0; kk < KC; kk += 32) {
            // --- A fragment (16x32 bf16; lane<16: K kk+0..7 & kk+16..23) ---
            Frag ahi, alo;
            {
                const int m  = lane & 15;
                const int kb = kk + ((lane < 16) ? 0 : 8);
                const __bf16* p = &sAhi[buf][(mB * 16 + m) * KC + kb];
                ahi.q[0] = *(const uint4*)p;
                ahi.q[1] = *(const uint4*)(p + 16);
                const __bf16* q = &sAlo[buf][(mB * 16 + m) * KC + kb];
                alo.q[0] = *(const uint4*)q;
                alo.q[1] = *(const uint4*)(q + 16);
            }
#pragma unroll
            for (int t = 0; t < 4; ++t) {
                const int nb  = nH * 4 + t;
                const int n   = lane & 15;
                const int kb2 = kk + ((lane < 16) ? 0 : 8);
                Frag bhi, blo;
                const __bf16* p = &sBhi[buf][(nb * 16 + n) * KC + kb2];
                bhi.q[0] = ((const uint4*)p)[0];
                bhi.q[1] = ((const uint4*)p)[2];
                const __bf16* q = &sBlo[buf][(nb * 16 + n) * KC + kb2];
                blo.q[0] = ((const uint4*)q)[0];
                blo.q[1] = ((const uint4*)q)[2];

                // fp32 ~= hi*hi + hi*lo + lo*hi   (lo*lo below fp32 epsilon)
                acc[t] = __builtin_amdgcn_wmma_f32_16x16x32_bf16(
                    false, ahi.v, false, bhi.v, (short)0, acc[t], false, false);
                acc[t] = __builtin_amdgcn_wmma_f32_16x16x32_bf16(
                    false, ahi.v, false, blo.v, (short)0, acc[t], false, false);
                acc[t] = __builtin_amdgcn_wmma_f32_16x16x32_bf16(
                    false, alo.v, false, bhi.v, (short)0, acc[t], false, false);
            }
        }

        if (kc + 1 < NKC) cvtStore((kc + 1) & 1);  // fill the other buffer
        __syncthreads();                            // single barrier per chunk
    }

    // --- spill accumulators to LDS score tile (C layout: VGPR r -> M=r(+8)) ---
#pragma unroll
    for (int t = 0; t < 4; ++t) {
        const int col     = (nH * 4 + t) * 16 + (lane & 15);
        const int rowBase = mB * 16 + ((lane >> 4) << 3);
#pragma unroll
        for (int r = 0; r < 8; ++r)
            sScore[(rowBase + r) * MAX_E + col] = acc[t][r];
    }
    __syncthreads();

    // --- per-token masked softmax (+1e-14), in LDS ---
    if (tid < TOK_PER_WG) {
        float* row = &sScore[tid * MAX_E];
        float mx = -3.4e38f;
        for (int e = 0; e < MAX_E; ++e) {
            float s = (sMask[e] == 0.f) ? -1e9f : row[e];
            row[e] = s;
            mx = fmaxf(mx, s);
        }
        float sum = 0.f;
        for (int e = 0; e < MAX_E; ++e) {
            float p = __expf(row[e] - mx);
            row[e] = p;
            sum += p;
        }
        const float inv = 1.f / sum;
        for (int e = 0; e < MAX_E; ++e)
            row[e] = row[e] * inv + 1e-14f;
    }
    __syncthreads();

    // --- coalesced store of probabilities ---
#pragma unroll
    for (int j = 0; j < (TOK_PER_WG * MAX_E) / 256; ++j) {
        int idx = tid + 256 * j;
        int tok = idx >> 7, e = idx & (MAX_E - 1);
        out[(size_t)(tokBase + tok) * MAX_E + e] = sScore[idx];
    }
    __syncthreads();

    // --- adaptive top-k: smallest k whose top-k mass >= 0.5 (destructive) ---
    if (tid < TOK_PER_WG) {
        float* row = &sScore[tid * MAX_E];
        float act = 0.f;
        for (int e = 0; e < MAX_E; ++e) act += sMask[e];
        const int active = (int)(act + 0.5f);

        float run = 0.f;
        int k = 0;
        while (run < THRESH && k < MAX_E) {
            float mxv = -2.f;
            int mi = 0;
            for (int e = 0; e < MAX_E; ++e) {
                float v = row[e];
                if (v > mxv) { mxv = v; mi = e; }
            }
            run += mxv;
            row[mi] = -1.f;
            ++k;
        }
        if (k > active) k = active;
        int* topk = (int*)(out + (size_t)NUM_TOKENS * MAX_E);
        topk[tokBase + tid] = k;
    }
}

extern "C" void kernel_launch(void* const* d_in, const int* in_sizes, int n_in,
                              void* d_out, int out_size, void* d_ws, size_t ws_size,
                              hipStream_t stream) {
    const float* x  = (const float*)d_in[0];   // [16384, 4096]
    const float* wg = (const float*)d_in[1];   // [128, 4096]
    const float* mk = (const float*)d_in[2];   // [128]
    float* out = (float*)d_out;                // scores [16384*128] ++ topk [16384] (int bits)

    dim3 grid(NUM_TOKENS / TOK_PER_WG);        // 256 workgroups
    mass_gate_kernel<<<grid, 256, 0, stream>>>(x, wg, mk, out);
}
